// MultiheadAttention_54898271978065
// MI455X (gfx1250) — compile-verified, matrix operand order checked by reading, not run on hardware
//
#include <hip/hip_runtime.h>
#include <hip/hip_bf16.h>

// ---------------------------------------------------------------------------
// Multihead attention for MI455X (gfx1250, wave32, WMMA f16 -> f32 accum)
// B=2, S=4096, D=768, H=12, HD=64
// Attention stage stages K/V chunks into LDS with the Tensor Data Mover
// (double-buffered, s_wait_tensorcnt), P round-trips through LDS for the
// C-fragment -> A-fragment relayout.
// ---------------------------------------------------------------------------

#define BATCH  2
#define SEQ    4096
#define DMODEL 768
#define NH     12
#define HDIM   64
#define MTOT   (BATCH * SEQ)          // 8192 rows
#define QKV_N  (3 * DMODEL)           // 2304
#define ATT_SCALE 0.125f              // 64^-0.5

typedef __attribute__((ext_vector_type(16))) _Float16 v16h;
typedef __attribute__((ext_vector_type(8)))  _Float16 v8h;
typedef __attribute__((ext_vector_type(8)))  float    v8f;
typedef unsigned int v4u __attribute__((ext_vector_type(4)));
typedef int          v4i __attribute__((ext_vector_type(4)));
typedef int          v8i __attribute__((ext_vector_type(8)));

union V16U { v16h v; v8h h[2]; };

// Load a 16x32 f16 fragment (A-style or "B^T"-style: row index = lane's m or n,
// contraction dim K along the row).  Per the CDNA5 16-bit A-matrix layout:
//   lanes 0-15  hold K = [0..7]  (VGPR0-3) and K = [16..23] (VGPR4-7)
//   lanes 16-31 hold K = [8..15] and K = [24..31]
// -> two 16-byte contiguous loads per lane (global_load_b128 or ds_load_b128).
__device__ __forceinline__ v16h load_frag16x32(const _Float16* base, int row,
                                               int lda, int k0, int hi) {
  const _Float16* p = base + (size_t)row * (size_t)lda + k0 + hi * 8;
  V16U u;
  u.h[0] = *(const v8h*)(p);
  u.h[1] = *(const v8h*)(p + 16);
  return u.v;
}

__device__ __forceinline__ v8f wmma16(v16h a, v16h b, v8f c) {
  return __builtin_amdgcn_wmma_f32_16x16x32_f16(
      /*neg_a=*/false, a, /*neg_b=*/false, b,
      /*c_mod=*/(short)0, c, /*reuse_a=*/false, /*reuse_b=*/false);
}

// ---------------------------------------------------------------------------
// Tensor Data Mover helpers
// ---------------------------------------------------------------------------
#if __has_builtin(__builtin_amdgcn_tensor_load_to_lds)
#define HAVE_TDM 1
#else
#define HAVE_TDM 0
#endif

// Generic pointers into LDS carry the LDS byte offset in their low 32 bits
// (aperture check discards the upper bits), so this gives D#.lds_addr.
__device__ __forceinline__ unsigned lds_addr_u32(const void* p) {
  return (unsigned)(unsigned long long)p;
}

#if HAVE_TDM
// 2D tile DMA: tile_y rows of tile_x elements (f16), row stride stride_x
// elements, packed contiguously into LDS at ldsaddr.  D# layout per CDNA5 ISA
// 8.3/8.4 (group0: count|lds_addr|global_addr|type=2; group1: data_size=2B,
// tensor dims, tile dims, dim0 stride).
__device__ __forceinline__ void tdm_load_2d(const void* gsrc, unsigned ldsaddr,
                                            unsigned tile_x, unsigned tile_y,
                                            unsigned tensor_x, unsigned tensor_y,
                                            unsigned long long stride_x) {
  unsigned long long ga = (unsigned long long)gsrc;
  v4u g0;
  g0.x = 1u;                                    // count=1 (valid user D#)
  g0.y = ldsaddr;                               // lds_addr (bytes)
  g0.z = (unsigned)ga;                          // global_addr[31:0]
  g0.w = (unsigned)((ga >> 32) & 0x1FFFFFFull)  // global_addr[56:32]
         | (2u << 30);                          // type = 2 ("image")
  v8i g1;
  g1[0] = (int)(1u << 16);                      // workgroup_mask=0, data_size=2B
  g1[1] = (int)((tensor_x & 0xFFFFu) << 16);    // tensor_dim0[15:0] @ bits63:48
  g1[2] = (int)((tensor_x >> 16) | ((tensor_y & 0xFFFFu) << 16));
  g1[3] = (int)((tensor_y >> 16) | ((tile_x & 0xFFFFu) << 16)); // tile_dim0
  g1[4] = (int)(tile_y & 0xFFFFu);              // tile_dim1 (tile_dim2 = 0)
  g1[5] = (int)(stride_x & 0xFFFFFFFFull);      // tensor_dim0_stride[31:0]
  g1[6] = (int)((stride_x >> 32) & 0xFFFFull);  // dim0_stride[47:32], dim1_stride=0
  g1[7] = 0;
  v4i z = {0, 0, 0, 0};
#if defined(__clang_major__) && __clang_major__ >= 23
  v8i z8 = {0, 0, 0, 0, 0, 0, 0, 0};
  __builtin_amdgcn_tensor_load_to_lds(g0, g1, z, z, z8, 0);
#else
  __builtin_amdgcn_tensor_load_to_lds(g0, g1, z, z, 0);
#endif
}
#endif // HAVE_TDM

// ---------------------------------------------------------------------------
// Stage 0: fp32 -> f16 cast
// ---------------------------------------------------------------------------
__global__ void cast_f32_to_f16(const float* __restrict__ src,
                                _Float16* __restrict__ dst, int n) {
  int i = blockIdx.x * blockDim.x + threadIdx.x;
  if (i < n) dst[i] = (_Float16)src[i];
}

// ---------------------------------------------------------------------------
// Stage 1: QKV projection  C[M=8192, N=2304] = Xh @ Wh^T + bias
// One wave computes a 16(M) x 64(N) tile.  Epilogue scatters into
// Qh [b,h,s,d] (pre-scaled), Kh [b,h,s,d], and Vt [b,h,d,s] (transposed).
// ---------------------------------------------------------------------------
__global__ __launch_bounds__(128) void qkv_gemm_kernel(
    const _Float16* __restrict__ xh,   // [8192, 768]
    const _Float16* __restrict__ wh,   // [2304, 768]
    const float*    __restrict__ bias, // [2304]
    _Float16* __restrict__ Qh,         // [B*H, S, HD]
    _Float16* __restrict__ Kh,         // [B*H, S, HD]
    _Float16* __restrict__ Vt) {       // [B*H, HD, S]
  const int wave = threadIdx.x >> 5;
  const int lane = threadIdx.x & 31;
  const int hi   = lane >> 4;
  const int lr   = lane & 15;

  const int MT = MTOT / 16;                 // 512
  int tile = blockIdx.x * 4 + wave;         // 512*36 = 18432 tiles
  int mt = tile % MT;
  int nt = tile / MT;
  int m0 = mt * 16;
  int n0 = nt * 64;

  v8f acc[4] = {};
  for (int k0 = 0; k0 < DMODEL; k0 += 32) {
    if (k0 + 32 < DMODEL)
      __builtin_prefetch(xh + (size_t)(m0 + lr) * DMODEL + k0 + 32, 0, 3);
    v16h a = load_frag16x32(xh, m0 + lr, DMODEL, k0, hi);
#pragma unroll
    for (int j = 0; j < 4; ++j) {
      v16h b = load_frag16x32(wh, n0 + j * 16 + lr, DMODEL, k0, hi);
      acc[j] = wmma16(a, b, acc[j]);
    }
  }

#pragma unroll
  for (int j = 0; j < 4; ++j) {
    int n = n0 + j * 16 + lr;
    float bi = bias[n];
    int region = n / DMODEL;    // 0 = Q, 1 = K, 2 = V
    int nn = n % DMODEL;
    int h = nn >> 6;
    int d = nn & 63;
#pragma unroll
    for (int r = 0; r < 8; ++r) {
      int m = m0 + r + hi * 8;
      int b = m >> 12;
      int s = m & (SEQ - 1);
      int bh = b * NH + h;
      float v = acc[j][r] + bi;
      if (region == 0) {
        Qh[((size_t)bh * SEQ + s) * HDIM + d] = (_Float16)(v * ATT_SCALE);
      } else if (region == 1) {
        Kh[((size_t)bh * SEQ + s) * HDIM + d] = (_Float16)v;
      } else {
        Vt[((size_t)bh * HDIM + d) * SEQ + s] = (_Float16)v;
      }
    }
  }
}

// ---------------------------------------------------------------------------
// Stage 2: flash attention.  A block's 4 waves share one (b,h) and walk the
// same key chunks, so K/V chunks are TDM-staged into LDS once per block
// (double-buffered).  Each wave owns 16 queries with online softmax.
// ---------------------------------------------------------------------------
__global__ __launch_bounds__(128) void attn_kernel(
    const _Float16* __restrict__ Qh,   // [B*H, S, HD], pre-scaled
    const _Float16* __restrict__ Kh,   // [B*H, S, HD]
    const _Float16* __restrict__ Vt,   // [B*H, HD, S]
    _Float16* __restrict__ Oh) {       // [B*S, D] row-major
  __shared__ __align__(16) _Float16 Kbuf[2][32 * HDIM];  // [key_local][d]
  __shared__ __align__(16) _Float16 Vbuf[2][HDIM * 32];  // [d][key_local]
  __shared__ __align__(16) _Float16 Plds[4][16 * 32];

  const int wave = threadIdx.x >> 5;
  const int lane = threadIdx.x & 31;
  const int hi   = lane >> 4;
  const int lr   = lane & 15;

  int tile = blockIdx.x * 4 + wave;     // B*H*(S/16) = 6144 tiles
  int bh = tile >> 8;                   // 4 waves of a block share bh
  int q0 = (tile & 255) * 16;

  const _Float16* Qb = Qh + (size_t)bh * SEQ * HDIM;
  const _Float16* Kb = Kh + (size_t)bh * SEQ * HDIM;
  const _Float16* Vb = Vt + (size_t)bh * HDIM * SEQ;
  _Float16* P = Plds[wave];

  // Q fragments for k = [0,32) and [32,64): resident for the whole key loop.
  v16h qa0 = load_frag16x32(Qb, q0 + lr, HDIM, 0, hi);
  v16h qa1 = load_frag16x32(Qb, q0 + lr, HDIM, 32, hi);

  v8f o[4] = {};
  float mrow[8], lrow[8];
#pragma unroll
  for (int r = 0; r < 8; ++r) { mrow[r] = -INFINITY; lrow[r] = 0.0f; }

#if HAVE_TDM
  if (wave == 0) {
    // preload chunk 0 into buffer 0
    tdm_load_2d(Kb, lds_addr_u32(Kbuf[0]), HDIM, 32, HDIM, SEQ, HDIM);
    tdm_load_2d(Vb, lds_addr_u32(Vbuf[0]), 32, HDIM, SEQ, HDIM, SEQ);
  }
#endif

  for (int j = 0; j < SEQ; j += 32) {
    int cur = (j >> 5) & 1;
    const _Float16* Kt = Kbuf[cur];
    const _Float16* Vc = Vbuf[cur];

#if HAVE_TDM
    if (wave == 0) {
      if (j + 32 < SEQ) {
        // issue next chunk, then drain down to it: current chunk is ready.
        tdm_load_2d(Kb + (size_t)(j + 32) * HDIM, lds_addr_u32(Kbuf[cur ^ 1]),
                    HDIM, 32, HDIM, SEQ, HDIM);
        tdm_load_2d(Vb + (j + 32), lds_addr_u32(Vbuf[cur ^ 1]),
                    32, HDIM, SEQ, HDIM, SEQ);
        __builtin_amdgcn_s_wait_tensorcnt((short)2);
      } else {
        __builtin_amdgcn_s_wait_tensorcnt((short)0);
      }
    }
    __syncthreads();
#else
    // Fallback: cooperative synchronous staging (single buffer).
    Kt = Kbuf[0];
    Vc = Vbuf[0];
    __syncthreads();
    {
      int t = threadIdx.x;
#pragma unroll
      for (int i = 0; i < 2; ++i) {        // K: 32 rows x 8 b128 = 256 chunks
        int idx = t + i * 128;
        int row = idx >> 3, c = idx & 7;
        *(v8h*)(Kbuf[0] + row * HDIM + c * 8) =
            *(const v8h*)(Kb + (size_t)(j + row) * HDIM + c * 8);
      }
#pragma unroll
      for (int i = 0; i < 2; ++i) {        // V: 64 rows x 4 b128 = 256 chunks
        int idx = t + i * 128;
        int row = idx >> 2, c = idx & 3;
        *(v8h*)(Vbuf[0] + row * 32 + c * 8) =
            *(const v8h*)(Vb + (size_t)row * SEQ + j + c * 8);
      }
    }
    __syncthreads();
#endif

    // ---- scores: S[16q x 32k] = Q @ K^T (K fragments from LDS) ----
    v8f s0 = {}, s1 = {};
    {
      v16h kb = load_frag16x32(Kt, lr, HDIM, 0, hi);
      s0 = wmma16(qa0, kb, s0);
      kb = load_frag16x32(Kt, lr, HDIM, 32, hi);
      s0 = wmma16(qa1, kb, s0);
      kb = load_frag16x32(Kt, 16 + lr, HDIM, 0, hi);
      s1 = wmma16(qa0, kb, s1);
      kb = load_frag16x32(Kt, 16 + lr, HDIM, 32, hi);
      s1 = wmma16(qa1, kb, s1);
    }

    // ---- online softmax (rows live at m = r + hi*8; reduce over 16 lanes) --
    float corr[8];
#pragma unroll
    for (int r = 0; r < 8; ++r) {
      float mx = fmaxf(s0[r], s1[r]);
#pragma unroll
      for (int off = 1; off < 16; off <<= 1)
        mx = fmaxf(mx, __shfl_xor(mx, off, 32));
      float mn = fmaxf(mrow[r], mx);
      float c  = __expf(mrow[r] - mn);
      float p0 = __expf(s0[r] - mn);
      float p1 = __expf(s1[r] - mn);
      int mloc = r + hi * 8;
      P[mloc * 32 + lr]      = (_Float16)p0;
      P[mloc * 32 + 16 + lr] = (_Float16)p1;
      float ps = p0 + p1;
#pragma unroll
      for (int off = 1; off < 16; off <<= 1)
        ps += __shfl_xor(ps, off, 32);
      lrow[r] = lrow[r] * c + ps;
      mrow[r] = mn;
      corr[r] = c;
    }

    // ---- rescale running output ----
#pragma unroll
    for (int nt = 0; nt < 4; ++nt)
#pragma unroll
      for (int r = 0; r < 8; ++r) o[nt][r] = o[nt][r] * corr[r];

    // ---- PV: O += P[16x32] @ V[32k x 64d] (both operands from LDS) ----
    v16h pa = load_frag16x32(P, lr, 32, 0, hi);
#pragma unroll
    for (int nt = 0; nt < 4; ++nt) {
      v16h vb = load_frag16x32(Vc, nt * 16 + lr, 32, 0, hi);
      o[nt] = wmma16(pa, vb, o[nt]);
    }

    __syncthreads();   // all waves done with buf[cur] before it is re-filled
  }

  // ---- normalize and store f16 for the projection GEMM ----
  int b = bh / NH;
  int h = bh % NH;
#pragma unroll
  for (int r = 0; r < 8; ++r) {
    float inv = 1.0f / lrow[r];
    int s = q0 + r + hi * 8;
    size_t row = ((size_t)b * SEQ + s) * DMODEL + h * HDIM;
#pragma unroll
    for (int nt = 0; nt < 4; ++nt) {
      Oh[row + nt * 16 + lr] = (_Float16)(o[nt][r] * inv);
    }
  }
}

// ---------------------------------------------------------------------------
// Stage 3: output projection  out[M=8192, N=768] = Oh @ Pw^T + bias (fp32 out)
// ---------------------------------------------------------------------------
__global__ __launch_bounds__(128) void proj_gemm_kernel(
    const _Float16* __restrict__ ah,   // [8192, 768]
    const _Float16* __restrict__ wh,   // [768, 768]
    const float*    __restrict__ bias, // [768]
    float* __restrict__ out) {         // [8192, 768]
  const int wave = threadIdx.x >> 5;
  const int lane = threadIdx.x & 31;
  const int hi   = lane >> 4;
  const int lr   = lane & 15;

  const int MT = MTOT / 16;                 // 512
  int tile = blockIdx.x * 4 + wave;         // 512*12 = 6144 tiles
  int mt = tile % MT;
  int nt = tile / MT;
  int m0 = mt * 16;
  int n0 = nt * 64;

  v8f acc[4] = {};
  for (int k0 = 0; k0 < DMODEL; k0 += 32) {
    if (k0 + 32 < DMODEL)
      __builtin_prefetch(ah + (size_t)(m0 + lr) * DMODEL + k0 + 32, 0, 3);
    v16h a = load_frag16x32(ah, m0 + lr, DMODEL, k0, hi);
#pragma unroll
    for (int j = 0; j < 4; ++j) {
      v16h b = load_frag16x32(wh, n0 + j * 16 + lr, DMODEL, k0, hi);
      acc[j] = wmma16(a, b, acc[j]);
    }
  }

#pragma unroll
  for (int j = 0; j < 4; ++j) {
    int n = n0 + j * 16 + lr;
    float bi = bias[n];
#pragma unroll
    for (int r = 0; r < 8; ++r) {
      int m = m0 + r + hi * 8;
      out[(size_t)m * DMODEL + n] = acc[j][r] + bi;
    }
  }
}

// ---------------------------------------------------------------------------
// Launcher
// ---------------------------------------------------------------------------
extern "C" void kernel_launch(void* const* d_in, const int* in_sizes, int n_in,
                              void* d_out, int out_size, void* d_ws, size_t ws_size,
                              hipStream_t stream) {
  (void)in_sizes; (void)n_in; (void)out_size; (void)ws_size;

  const float* x      = (const float*)d_in[0];   // [B,S,D]
  const float* qkv_w  = (const float*)d_in[1];   // [3D, D]
  const float* qkv_b  = (const float*)d_in[2];   // [3D]
  const float* proj_w = (const float*)d_in[3];   // [D, D]
  const float* proj_b = (const float*)d_in[4];   // [D]
  float* out = (float*)d_out;

  const size_t N_XH   = (size_t)MTOT * DMODEL;            // 6291456
  const size_t N_QW   = (size_t)QKV_N * DMODEL;           // 1769472
  const size_t N_PW   = (size_t)DMODEL * DMODEL;          // 589824
  const size_t N_HEAD = (size_t)BATCH * NH * SEQ * HDIM;  // 6291456

  auto align256 = [](size_t b) { return (b + 255) & ~(size_t)255; };
  char* ws = (char*)d_ws;
  size_t off = 0;
  _Float16* xh  = (_Float16*)(ws + off); off = align256(off + N_XH * 2);
  _Float16* qwh = (_Float16*)(ws + off); off = align256(off + N_QW * 2);
  _Float16* pwh = (_Float16*)(ws + off); off = align256(off + N_PW * 2);
  _Float16* Qh  = (_Float16*)(ws + off); off = align256(off + N_HEAD * 2);
  _Float16* Kh  = (_Float16*)(ws + off); off = align256(off + N_HEAD * 2);
  _Float16* Vt  = (_Float16*)(ws + off); off = align256(off + N_HEAD * 2);
  _Float16* Oh  = (_Float16*)(ws + off); off = align256(off + N_XH * 2);

  {
    int n0 = (int)N_XH, n1 = (int)N_QW, n2 = (int)N_PW;
    cast_f32_to_f16<<<(n0 + 255) / 256, 256, 0, stream>>>(x, xh, n0);
    cast_f32_to_f16<<<(n1 + 255) / 256, 256, 0, stream>>>(qkv_w, qwh, n1);
    cast_f32_to_f16<<<(n2 + 255) / 256, 256, 0, stream>>>(proj_w, pwh, n2);
  }

  qkv_gemm_kernel<<<(MTOT / 16) * (QKV_N / 64) / 4, 128, 0, stream>>>(
      xh, qwh, qkv_b, Qh, Kh, Vt);

  attn_kernel<<<BATCH * NH * (SEQ / 16) / 4, 128, 0, stream>>>(Qh, Kh, Vt, Oh);

  proj_gemm_kernel<<<(MTOT / 16) * (DMODEL / 64) / 4, 128, 0, stream>>>(
      Oh, pwh, proj_b, out);
}
